// TGATUNet_49134425866410
// MI455X (gfx1250) — matrix operand order repaired and checked by reading, not compiled, not run on hardware
//
#include <hip/hip_runtime.h>
#include <hip/hip_bf16.h>

// ---------------------------------------------------------------------------
// TGAT-UNet for MI455X (gfx1250). All dense math via v_wmma_f32_16x16x32_bf16.
// Problem is L2-resident (~6MB live data vs 192MB L2) => compute bound; WMMA
// bf16 with f32 accumulation is the throughput path on CDNA5.
// GEMM uses 32x32 per-wave register blocking: 4 WMMAs per 6 fragment loads
// (vs 1 per 4 unblocked) to keep the matrix pipe ahead of the vmem pipe.
// ---------------------------------------------------------------------------

typedef __bf16 bf16_t;
typedef __attribute__((ext_vector_type(16))) __bf16 v16bf;
typedef __attribute__((ext_vector_type(8)))  __bf16 v8bf;
typedef __attribute__((ext_vector_type(8)))  float  v8f;

#define NTOK   4096
#define DIN    64
#define HDIM   256
#define NHEADS 4
#define DH     64
#define DFF    512
#define TIMEK  4

static __device__ __forceinline__ v8f wmma_bf16(v16bf a, v16bf b, v8f c) {
  // D = A(16x32 bf16) * B(32x16 bf16) + C(16x16 f32)
  return __builtin_amdgcn_wmma_f32_16x16x32_bf16(false, a, false, b, (short)0, c,
                                                 false, false);
}

// A fragment 16x32: lane row = lm, elems 0..15 -> K=kh*16+j
static __device__ __forceinline__ v16bf load_afrag(const bf16_t* rowp, int kh) {
  v8bf lo = *(const v8bf*)(rowp + kh * 16);
  v8bf hi = *(const v8bf*)(rowp + kh * 16 + 8);
  v16bf a;
#pragma unroll
  for (int j = 0; j < 8; ++j) { a[j] = lo[j]; a[8 + j] = hi[j]; }
  return a;
}

// ---------------------------------------------------------------------------
// Register-blocked WMMA GEMM: C[M,N] = act(A[M,K]*B[K,N] + bias + residual)
// A bf16 row-major, B pre-transposed bf16 (Bt[N][K]). 32x32 tile per wave.
// Optional f32 out, optional bf16 out (row-major or transposed [N][M]).
// ---------------------------------------------------------------------------
__global__ void gemm_wmma(const bf16_t* __restrict__ A, const bf16_t* __restrict__ Bt,
                          const float* __restrict__ bias, const float* __restrict__ res,
                          float* __restrict__ Cf, bf16_t* __restrict__ Cb,
                          int M, int N, int K, int relu, int transOut) {
  const int l  = threadIdx.x;          // lane in wave32
  const int lm = l & 15;
  const int kh = l >> 4;
  const int nbase = blockIdx.x * 32;
  const int mbase = (blockIdx.y * blockDim.y + threadIdx.y) * 32;

  const bf16_t* arow0 = A  + (size_t)(mbase + lm) * K;
  const bf16_t* arow1 = A  + (size_t)(mbase + 16 + lm) * K;
  const bf16_t* brow0 = Bt + (size_t)(nbase + lm) * K;
  const bf16_t* brow1 = Bt + (size_t)(nbase + 16 + lm) * K;

  v8f c00 = {}, c01 = {}, c10 = {}, c11 = {};
  for (int k = 0; k < K; k += 32) {
    v16bf a0 = load_afrag(arow0 + k, kh);
    v16bf a1 = load_afrag(arow1 + k, kh);
    // B fragment 32x16: lane col = lm, elem i -> K = k + kh*16 + i (contig in Bt)
    v16bf b0 = *(const v16bf*)(brow0 + k + kh * 16);
    v16bf b1 = *(const v16bf*)(brow1 + k + kh * 16);
    c00 = wmma_bf16(a0, b0, c00);
    c01 = wmma_bf16(a0, b1, c01);
    c10 = wmma_bf16(a1, b0, c10);
    c11 = wmma_bf16(a1, b1, c11);
  }

  v8f acc[2][2] = {{c00, c01}, {c10, c11}};
#pragma unroll
  for (int mt = 0; mt < 2; ++mt) {
#pragma unroll
    for (int nt = 0; nt < 2; ++nt) {
      const int n  = nbase + nt * 16 + lm;
      const float bv = bias ? bias[n] : 0.0f;
#pragma unroll
      for (int r = 0; r < 8; ++r) {
        const int m = mbase + mt * 16 + r + 8 * kh;   // C layout: M = r + 8*(lane>>4)
        float val = acc[mt][nt][r] + bv;
        if (res)  val += res[(size_t)m * N + n];
        if (relu) val = fmaxf(val, 0.0f);
        if (Cf) Cf[(size_t)m * N + n] = val;
        if (Cb) {
          if (transOut) Cb[(size_t)n * M + m] = (bf16_t)val;
          else          Cb[(size_t)m * N + n] = (bf16_t)val;
        }
      }
    }
  }
}

// ---------------------------------------------------------------------------
// Flash attention, one wave per (16-query tile, head). dh=64, scale=0.125.
// Q,K bf16 row-major [NTOK][256]; V pre-transposed bf16 [256][NTOK].
// Output bf16 row-major [NTOK][256]. Online softmax in C-fragment layout,
// P tile staged through LDS to reshape into the A-fragment layout.
// ---------------------------------------------------------------------------
__global__ void flash_attn(const bf16_t* __restrict__ Qb, const bf16_t* __restrict__ Kb,
                           const bf16_t* __restrict__ Vt, bf16_t* __restrict__ Ob) {
  __shared__ __align__(32) bf16_t Pt[16][32];

  const int l  = threadIdx.x;
  const int lm = l & 15;
  const int kh = l >> 4;
  const int qbase = blockIdx.x * 16;
  const int h     = blockIdx.y;

  // Q fragments for contraction dh 0..31 and 32..63 (loaded once, reused 256x)
  const bf16_t* qrow = Qb + (size_t)(qbase + lm) * HDIM + h * DH;
  v16bf qa0 = load_afrag(qrow, kh);
  v16bf qa1 = load_afrag(qrow + 32, kh);

  float rowmax[8], rowsum[8];
  v8f o[4] = {};
#pragma unroll
  for (int r = 0; r < 8; ++r) { rowmax[r] = -1e30f; rowsum[r] = 0.0f; }

  for (int cb = 0; cb < NTOK; cb += 32) {
    // ---- scores S = (Q * K^T) * 0.125 for 32 keys (two 16x16 tiles) ----
    const bf16_t* kr0 = Kb + (size_t)(cb + lm) * HDIM + h * DH + kh * 16;
    const bf16_t* kr1 = Kb + (size_t)(cb + 16 + lm) * HDIM + h * DH + kh * 16;
    v16bf kb00 = *(const v16bf*)(kr0);        // dh = kh*16+i
    v16bf kb01 = *(const v16bf*)(kr0 + 32);   // dh = 32+kh*16+i
    v16bf kb10 = *(const v16bf*)(kr1);
    v16bf kb11 = *(const v16bf*)(kr1 + 32);
    v8f s0 = {}, s1 = {};
    s0 = wmma_bf16(qa0, kb00, s0);
    s0 = wmma_bf16(qa1, kb01, s0);
    s1 = wmma_bf16(qa0, kb10, s1);
    s1 = wmma_bf16(qa1, kb11, s1);

    // ---- online softmax per row (rows live across 16-lane halves) ----
#pragma unroll
    for (int r = 0; r < 8; ++r) {
      float a0 = s0[r] * 0.125f;
      float a1 = s1[r] * 0.125f;
      float mx = fmaxf(a0, a1);
#pragma unroll
      for (int off = 8; off > 0; off >>= 1) mx = fmaxf(mx, __shfl_xor(mx, off, 16));
      float nm   = fmaxf(rowmax[r], mx);
      float corr = __expf(rowmax[r] - nm);
      float p0 = __expf(a0 - nm);
      float p1 = __expf(a1 - nm);
      float ps = p0 + p1;
#pragma unroll
      for (int off = 8; off > 0; off >>= 1) ps += __shfl_xor(ps, off, 16);
      rowsum[r] = rowsum[r] * corr + ps;
      rowmax[r] = nm;
#pragma unroll
      for (int t = 0; t < 4; ++t) o[t][r] *= corr;
      const int m = r + 8 * kh;
      Pt[m][lm]      = (bf16_t)p0;
      Pt[m][16 + lm] = (bf16_t)p1;
    }
    __syncthreads();

    // ---- reload P as A-fragment (16x32) ----
    v16bf pa = load_afrag(&Pt[lm][0], kh);

    // ---- O += P * V  (V pre-transposed -> contiguous B fragments) ----
#pragma unroll
    for (int t = 0; t < 4; ++t) {
      const bf16_t* vr = Vt + (size_t)(h * DH + t * 16 + lm) * NTOK + cb + kh * 16;
      v16bf vb = *(const v16bf*)(vr);
      o[t] = wmma_bf16(pa, vb, o[t]);
    }
    __syncthreads();
  }

  // ---- normalize + store ----
#pragma unroll
  for (int t = 0; t < 4; ++t) {
#pragma unroll
    for (int r = 0; r < 8; ++r) {
      const int m = qbase + r + 8 * kh;
      const int n = h * DH + t * 16 + lm;
      Ob[(size_t)m * HDIM + n] = (bf16_t)(o[t][r] / rowsum[r]);
    }
  }
}

// ---------------------------------------------------------------------------
// Small scalar helper kernels (negligible FLOPs)
// ---------------------------------------------------------------------------
__global__ void f32_to_bf16(const float* __restrict__ x, bf16_t* __restrict__ y, int n) {
  int i = blockIdx.x * blockDim.x + threadIdx.x;
  if (i < n) y[i] = (bf16_t)x[i];
}

// Wt[n*K + k] = bf16(W[k*N + n])
__global__ void wtrans(const float* __restrict__ W, bf16_t* __restrict__ Wt, int K, int N) {
  int i = blockIdx.x * blockDim.x + threadIdx.x;
  if (i >= K * N) return;
  int k = i / N, n = i % N;
  Wt[(size_t)n * K + k] = (bf16_t)W[i];
}

// es/ed = sum_d h[n,hd,d] * a_{src,dst}[hd,d]
__global__ void gat_coef(const float* __restrict__ h, const float* __restrict__ as,
                         const float* __restrict__ ad, float* __restrict__ es,
                         float* __restrict__ ed, int N, int heads, int dh) {
  int idx = blockIdx.x * blockDim.x + threadIdx.x;
  if (idx >= N * heads) return;
  int n = idx / heads, hd = idx % heads;
  const float* hp = h + (size_t)n * heads * dh + hd * dh;
  const float* ap = as + hd * dh;
  const float* bp = ad + hd * dh;
  float s = 0.0f, d = 0.0f;
  for (int i = 0; i < dh; ++i) { float hv = hp[i]; s += hv * ap[i]; d += hv * bp[i]; }
  es[idx] = s; ed[idx] = d;
}

// Band-structured GAT edge softmax + aggregate (neighbors j = i-4..i+4).
__global__ void gat_aggregate(const float* __restrict__ h, const float* __restrict__ es,
                              const float* __restrict__ ed, const float* __restrict__ bias,
                              float* __restrict__ outf, bf16_t* __restrict__ outbf,
                              int N, int heads, int dh, int relu) {
  int idx = blockIdx.x * blockDim.x + threadIdx.x;
  const int C = heads * dh;
  if (idx >= N * C) return;
  const int c  = idx % C;
  const int i  = idx / C;
  const int hd = c / dh;
  const int j0 = (i - TIMEK < 0) ? 0 : i - TIMEK;
  const int j1 = (i + TIMEK > N - 1) ? N - 1 : i + TIMEK;
  const float edi = ed[i * heads + hd];
  float m = -1e30f;
  for (int j = j0; j <= j1; ++j) {
    float e = es[j * heads + hd] + edi;
    e = (e > 0.0f) ? e : 0.2f * e;           // LeakyReLU(0.2)
    m = fmaxf(m, e);
  }
  float s = 0.0f, acc = 0.0f;
  for (int j = j0; j <= j1; ++j) {
    float e = es[j * heads + hd] + edi;
    e = (e > 0.0f) ? e : 0.2f * e;
    float p = __expf(e - m);
    s += p;
    acc += p * h[(size_t)j * C + c];
  }
  float val = acc / s + bias[c];
  if (relu) val = fmaxf(val, 0.0f);
  outf[(size_t)i * C + c] = val;
  if (outbf) outbf[(size_t)i * C + c] = (bf16_t)val;
}

// Post-LN over D=256, writes f32 + bf16.
__global__ void layernorm(const float* __restrict__ x, const float* __restrict__ g,
                          const float* __restrict__ b, float* __restrict__ yf,
                          bf16_t* __restrict__ ybf, int D) {
  __shared__ float red[256];
  const int row = blockIdx.x;
  const int t   = threadIdx.x;
  float v = x[(size_t)row * D + t];
  red[t] = v; __syncthreads();
  for (int s = 128; s > 0; s >>= 1) { if (t < s) red[t] += red[t + s]; __syncthreads(); }
  float mu = red[0] / D; __syncthreads();
  float d = v - mu;
  red[t] = d * d; __syncthreads();
  for (int s = 128; s > 0; s >>= 1) { if (t < s) red[t] += red[t + s]; __syncthreads(); }
  float var = red[0] / D;
  float out = d * rsqrtf(var + 1e-5f) * g[t] + b[t];
  yf [(size_t)row * D + t] = out;
  ybf[(size_t)row * D + t] = (bf16_t)out;
}

// out[c*N + n] = x[n*C + c]
__global__ void out_transpose(const float* __restrict__ x, float* __restrict__ out,
                              int N, int C) {
  int idx = blockIdx.x * blockDim.x + threadIdx.x;
  if (idx >= N * C) return;
  int n = idx / C, c = idx % C;
  out[(size_t)c * N + n] = x[idx];
}

// ---------------------------------------------------------------------------
// Host orchestration
// ---------------------------------------------------------------------------
static inline size_t alignup(size_t v) { return (v + 255) & ~(size_t)255; }

extern "C" void kernel_launch(void* const* d_in, const int* in_sizes, int n_in,
                              void* d_out, int out_size, void* d_ws, size_t ws_size,
                              hipStream_t stream) {
  (void)in_sizes; (void)n_in; (void)out_size; (void)ws_size;

  // -------- input map (setup_inputs dict insertion order, depth-first) -----
  int ix = 0;
  auto F = [&](int i) { return (const float*)d_in[i]; };
  const float* window = F(ix++);                       // [4096,64]
  struct GatP { const float *W, *as, *ad, *b; };
  GatP enc[3], dec[3];
  for (int i = 0; i < 3; ++i) { enc[i].W = F(ix++); enc[i].as = F(ix++); enc[i].ad = F(ix++); enc[i].b = F(ix++); }
  struct TP { const float *Wq,*bq,*Wk,*bk,*Wv,*bv,*Wo,*bo,*g1,*b1,*g2,*b2,*W1,*bff1,*W2,*bff2; };
  TP tp[2];
  for (int i = 0; i < 2; ++i) {
    tp[i].Wq = F(ix++); tp[i].bq = F(ix++); tp[i].Wk = F(ix++); tp[i].bk = F(ix++);
    tp[i].Wv = F(ix++); tp[i].bv = F(ix++); tp[i].Wo = F(ix++); tp[i].bo = F(ix++);
    tp[i].g1 = F(ix++); tp[i].b1 = F(ix++); tp[i].g2 = F(ix++); tp[i].b2 = F(ix++);
    tp[i].W1 = F(ix++); tp[i].bff1 = F(ix++); tp[i].W2 = F(ix++); tp[i].bff2 = F(ix++);
  }
  for (int i = 0; i < 3; ++i) { dec[i].W = F(ix++); dec[i].as = F(ix++); dec[i].ad = F(ix++); dec[i].b = F(ix++); }
  // src/dst edge arrays unused: the graph is the fixed +/-4 temporal band.

  // -------- workspace carve (~30 MB) --------------------------------------
  char* w = (char*)d_ws;
  size_t off = 0;
  auto carve = [&](size_t bytes) { void* p = w + off; off += alignup(bytes); return p; };
  float*  xf    = (float*) carve((size_t)NTOK * HDIM * 4);   // activations f32
  float*  x2f   = (float*) carve((size_t)NTOK * HDIM * 4);   // pre-LN buffer
  float*  hbuf  = (float*) carve((size_t)NTOK * HDIM * 4);   // GAT h
  bf16_t* xbf   = (bf16_t*)carve((size_t)NTOK * HDIM * 2);
  bf16_t* qbf   = (bf16_t*)carve((size_t)NTOK * HDIM * 2);
  bf16_t* kbf   = (bf16_t*)carve((size_t)NTOK * HDIM * 2);
  bf16_t* vtb   = (bf16_t*)carve((size_t)NTOK * HDIM * 2);   // V^T [256][4096]
  bf16_t* attbf = (bf16_t*)carve((size_t)NTOK * HDIM * 2);
  bf16_t* f1bf  = (bf16_t*)carve((size_t)NTOK * DFF  * 2);
  bf16_t* wt    = (bf16_t*)carve((size_t)DFF  * HDIM * 2);   // transposed weight slot
  float*  es    = (float*) carve((size_t)NTOK * NHEADS * 4);
  float*  ed    = (float*) carve((size_t)NTOK * NHEADS * 4);

  auto launch_gemm = [&](const bf16_t* A, const bf16_t* Bt, const float* bias,
                         const float* res, float* Cf, bf16_t* Cb,
                         int M, int N, int K, int relu, int transOut) {
    dim3 grid(N / 32, M / 128), blk(32, 4);   // 32x32 tile per wave, 4 waves/block
    gemm_wmma<<<grid, blk, 0, stream>>>(A, Bt, bias, res, Cf, Cb, M, N, K, relu, transOut);
  };
  auto launch_wt = [&](const float* W, int K, int N) {
    int tot = K * N;
    wtrans<<<(tot + 255) / 256, 256, 0, stream>>>(W, wt, K, N);
  };
  auto gat_layer = [&](const GatP& p, int din, int heads, int dh, int relu, bf16_t* obf) {
    int C = heads * dh;
    launch_wt(p.W, din, C);
    launch_gemm(xbf, wt, nullptr, nullptr, hbuf, nullptr, NTOK, C, din, 0, 0);
    int nc = NTOK * heads;
    gat_coef<<<(nc + 255) / 256, 256, 0, stream>>>(hbuf, p.as, p.ad, es, ed, NTOK, heads, dh);
    int tot = NTOK * C;
    gat_aggregate<<<(tot + 255) / 256, 256, 0, stream>>>(hbuf, es, ed, p.b, xf, obf,
                                                         NTOK, heads, dh, relu);
  };

  // -------- encoder --------------------------------------------------------
  f32_to_bf16<<<(NTOK * DIN + 255) / 256, 256, 0, stream>>>(window, xbf, NTOK * DIN);
  gat_layer(enc[0], DIN,  NHEADS, DH, 1, xbf);
  gat_layer(enc[1], HDIM, NHEADS, DH, 1, xbf);
  gat_layer(enc[2], HDIM, NHEADS, DH, 1, xbf);

  // -------- transformer layers ---------------------------------------------
  for (int t = 0; t < 2; ++t) {
    const TP& p = tp[t];
    launch_wt(p.Wq, HDIM, HDIM);
    launch_gemm(xbf, wt, p.bq, nullptr, nullptr, qbf, NTOK, HDIM, HDIM, 0, 0);
    launch_wt(p.Wk, HDIM, HDIM);
    launch_gemm(xbf, wt, p.bk, nullptr, nullptr, kbf, NTOK, HDIM, HDIM, 0, 0);
    launch_wt(p.Wv, HDIM, HDIM);
    launch_gemm(xbf, wt, p.bv, nullptr, nullptr, vtb, NTOK, HDIM, HDIM, 0, 1);  // V^T

    flash_attn<<<dim3(NTOK / 16, NHEADS), 32, 0, stream>>>(qbf, kbf, vtb, attbf);

    launch_wt(p.Wo, HDIM, HDIM);
    launch_gemm(attbf, wt, p.bo, xf, x2f, nullptr, NTOK, HDIM, HDIM, 0, 0);     // +residual
    layernorm<<<NTOK, HDIM, 0, stream>>>(x2f, p.g1, p.b1, xf, xbf, HDIM);

    launch_wt(p.W1, HDIM, DFF);
    launch_gemm(xbf, wt, p.bff1, nullptr, nullptr, f1bf, NTOK, DFF, HDIM, 1, 0); // relu
    launch_wt(p.W2, DFF, HDIM);
    launch_gemm(f1bf, wt, p.bff2, xf, x2f, nullptr, NTOK, HDIM, DFF, 0, 0);      // +residual
    layernorm<<<NTOK, HDIM, 0, stream>>>(x2f, p.g2, p.b2, xf, xbf, HDIM);
  }

  // -------- decoder --------------------------------------------------------
  gat_layer(dec[0], HDIM, NHEADS, DH, 1, xbf);
  gat_layer(dec[1], HDIM, NHEADS, DH, 1, xbf);
  gat_layer(dec[2], HDIM, 1, DIN, 0, nullptr);   // final: heads=1, out=64, no relu

  // -------- output: x.T [64, 4096] -----------------------------------------
  out_transpose<<<(NTOK * DIN + 255) / 256, 256, 0, stream>>>(xf, (float*)d_out, NTOK, DIN);
}
